// PooledProgressNet_16947940950589
// MI455X (gfx1250) — matrix operand order repaired and checked by reading, not compiled, not run on hardware
//
#include <hip/hip_runtime.h>
#include <hip/hip_bf16.h>
#include <math.h>

// ---------------------------------------------------------------------------
// PooledProgressNet on MI455X (gfx1250, wave32, WMMA)
//
//   1) spp_pool_kernel        frames -> pooled           (bandwidth bound, 226MB)
//   2) fc90_kernel            pooled -> fpooled          (tiny GEMM, VALU)
//   3) pack_* kernels         f32 -> f16 operand packing for WMMA
//   4) fused_mlp_kernel       [16384,96]x[96,4096]x[4096,64] fused,
//                             v_wmma_f32_16x16x32_f16, M-tile 32 (B reuse x2),
//                             async global->LDS A staging, hidden strip in LDS
//   5) lstm_seq_kernel        persistent single-WG sequential LSTM (S=512)
//   6) forecast_cell_kernel   batched one-step forecast LSTM (8192 rows)
// ---------------------------------------------------------------------------

typedef _Float16 half_t;
typedef __attribute__((ext_vector_type(16))) _Float16 v16h;
typedef __attribute__((ext_vector_type(8)))  _Float16 v8h;
typedef __attribute__((ext_vector_type(8)))  float    v8f;

#define BB 16
#define SS 512
#define NFRM 8192            // B*S
#define MROWS 16384          // 2*B*S (emb rows then femb rows)
#define KPAD 96              // 90 padded to 3*32
#define EE 4096
#define FRAME_ELEMS (3*48*48)

__device__ __forceinline__ float sigf(float x) { return 1.0f / (1.0f + expf(-x)); }

__device__ __forceinline__ v16h make_v16h(v8h lo, v8h hi) {
  return __builtin_shufflevector(lo, hi, 0,1,2,3,4,5,6,7,8,9,10,11,12,13,14,15);
}

// A fragment (16x32 f16, row-major source in LDS).
// Lane L: row = L&15; kbase = k0 + (L<16 ? 0 : 16); k=kbase+j, j=0..15.
// Two contiguous 16B LDS loads.
__device__ __forceinline__ v16h lds_afrag(const half_t* base, int stride, int k0, int lane) {
  int row = lane & 15;
  int kb  = k0 + ((lane & 16) ? 16 : 0);
  const half_t* p = base + row * stride + kb;
  v8h lo = *(const v8h*)(p);
  v8h hi = *(const v8h*)(p + 8);
  return make_v16h(lo, hi);
}

// B fragment (32x16 f16) from an N-major packed weight matrix Bt[N][kstride].
// Lane L: col = n0 + (L&15); k = k0 + (L<16 ? 0 : 16) + j, j=0..15 -> one
// contiguous 32B global load.
__device__ __forceinline__ v16h glb_bfrag(const half_t* Bt, int kstride, int n0, int k0, int lane) {
  int col = n0 + (lane & 15);
  int kb  = k0 + ((lane & 16) ? 16 : 0);
  return *(const v16h*)(Bt + (size_t)col * kstride + kb);
}

__device__ __forceinline__ v8f wmma32(v16h a, v16h b, v8f c) {
  return __builtin_amdgcn_wmma_f32_16x16x32_f16(false, a, false, b, (short)0, c, false, false);
}

// gfx1250 async global->LDS copy (ASYNCcnt-tracked, no VGPR round trip).
// VDST = per-lane LDS byte offset, VADDR = per-lane 64-bit global address.
__device__ __forceinline__ void async_copy_b128(const void* gptr, void* lptr) {
  unsigned lds_addr = (unsigned)(size_t)lptr;                 // low 32b of generic = LDS offset
  unsigned long long ga = (unsigned long long)(size_t)gptr;
  asm volatile("global_load_async_to_lds_b128 %0, %1, off" :: "v"(lds_addr), "v"(ga) : "memory");
}
__device__ __forceinline__ void wait_async0() {
  asm volatile("s_wait_asynccnt 0" ::: "memory");
}

// ---------------------------------------------------------------------------
// 1) SPP pooling: one block per frame; frame staged in LDS (27KB).
// Feature layout per frame: lvl4 ch-major 48, lvl3 27, lvl2 12, lvl1 3 (=90).
// ---------------------------------------------------------------------------
__global__ __launch_bounds__(128) void spp_pool_kernel(const float* __restrict__ frames,
                                                       float* __restrict__ pooled) {
  __shared__ float sF[FRAME_ELEMS];
  __shared__ float sL4[48];
  __shared__ float sL2[12];
  const int r = blockIdx.x, tid = threadIdx.x;
  const float* fr = frames + (size_t)r * FRAME_ELEMS;
  for (int i = tid; i < FRAME_ELEMS; i += 128) sF[i] = fr[i];
  __syncthreads();
  float* out = pooled + (size_t)r * 90;
  if (tid < 48) {                       // level 4: 12x12 cells
    int ch = tid >> 4, cell = tid & 15, cy = cell >> 2, cx = cell & 3;
    const float* base = sF + ch * 2304;
    float m = -3.402823466e38f;
    for (int y = cy * 12; y < cy * 12 + 12; ++y)
      for (int x = cx * 12; x < cx * 12 + 12; ++x)
        m = fmaxf(m, base[y * 48 + x]);
    sL4[tid] = m;
    out[ch * 16 + cell] = m;
  }
  if (tid >= 64 && tid < 91) {          // level 3: 16x16 cells (from raw)
    int t = tid - 64, ch = t / 9, cell = t % 9, cy = cell / 3, cx = cell % 3;
    const float* base = sF + ch * 2304;
    float m = -3.402823466e38f;
    for (int y = cy * 16; y < cy * 16 + 16; ++y)
      for (int x = cx * 16; x < cx * 16 + 16; ++x)
        m = fmaxf(m, base[y * 48 + x]);
    out[48 + ch * 9 + cell] = m;
  }
  __syncthreads();
  if (tid < 12) {                       // level 2 = max of 2x2 level-4 cells
    int ch = tid >> 2, cell = tid & 3, cy = cell >> 1, cx = cell & 1;
    float m = -3.402823466e38f;
    for (int dy = 0; dy < 2; ++dy)
      for (int dx = 0; dx < 2; ++dx)
        m = fmaxf(m, sL4[ch * 16 + (2 * cy + dy) * 4 + (2 * cx + dx)]);
    sL2[tid] = m;
    out[75 + ch * 4 + cell] = m;
  }
  __syncthreads();
  if (tid < 3) {                        // level 1 = max of 4 level-2 cells
    float m = -3.402823466e38f;
    for (int i = 0; i < 4; ++i) m = fmaxf(m, sL2[tid * 4 + i]);
    out[87 + tid] = m;
  }
}

// ---------------------------------------------------------------------------
// 2) fpooled = pooled @ forecast_fc_w.T + b   (90x90, tiny)
// ---------------------------------------------------------------------------
__global__ __launch_bounds__(96) void fc90_kernel(const float* __restrict__ pooled,
                                                  const float* __restrict__ w,
                                                  const float* __restrict__ b,
                                                  float* __restrict__ fpooled) {
  __shared__ float sx[90];
  const int r = blockIdx.x, tid = threadIdx.x;
  if (tid < 90) sx[tid] = pooled[(size_t)r * 90 + tid];
  __syncthreads();
  if (tid < 90) {
    float acc = b[tid];
    const float* wr = w + tid * 90;
    for (int k = 0; k < 90; ++k) acc += wr[k] * sx[k];
    fpooled[(size_t)r * 90 + tid] = acc;
  }
}

// ---------------------------------------------------------------------------
// 3) operand packing (f32 -> f16, K padded to 96, weights N-major)
// ---------------------------------------------------------------------------
__global__ __launch_bounds__(256) void pack_x_kernel(const float* __restrict__ pooled,
                                                     const float* __restrict__ fpooled,
                                                     half_t* __restrict__ Xh) {
  int idx = blockIdx.x * 256 + threadIdx.x;          // MROWS*KPAD threads
  if (idx >= MROWS * KPAD) return;
  int r = idx / KPAD, k = idx - r * KPAD;
  float v = 0.0f;
  if (k < 90) v = (r < NFRM) ? pooled[(size_t)r * 90 + k]
                             : fpooled[(size_t)(r - NFRM) * 90 + k];
  Xh[idx] = (half_t)v;
}

__global__ __launch_bounds__(256) void pack_w1_kernel(const float* __restrict__ w1,
                                                      half_t* __restrict__ B1t) {
  int idx = blockIdx.x * 256 + threadIdx.x;          // EE*KPAD threads
  if (idx >= EE * KPAD) return;
  int n = idx / KPAD, k = idx - n * KPAD;
  B1t[idx] = (half_t)((k < 90) ? w1[(size_t)n * 90 + k] : 0.0f);
}

__global__ __launch_bounds__(256) void pack_w2_kernel(const float* __restrict__ w2,
                                                      half_t* __restrict__ B2t) {
  int idx = blockIdx.x * 256 + threadIdx.x;          // 64*EE threads
  if (idx >= 64 * EE) return;
  B2t[idx] = (half_t)w2[idx];                        // fc7_w is already [64][4096] k-contiguous
}

// ---------------------------------------------------------------------------
// 4) Fused MLP:  emb = relu(relu(X @ W1.T + b1) @ W2.T + b2)
// One block per 32-row strip (two 16-row groups -> every B fragment feeds 2
// WMMAs, halving L2 weight traffic). Hidden strip (32x4096 f16) lives in LDS,
// processed in 8 chunks of 512 cols (32KB). A strip staged with
// global_load_async_to_lds_b128 (ASYNCcnt path).
// Stage 1: per chunk, 8 waves x 4 N-tiles x 3 k-steps x 2 row-groups.
// Stage 2: N=64 -> 4 tiles; wave = (ntile, khalf); partials reduced in LDS.
// ---------------------------------------------------------------------------
__global__ __launch_bounds__(256) void fused_mlp_kernel(const half_t* __restrict__ Xh,
                                                        const half_t* __restrict__ B1t,
                                                        const half_t* __restrict__ B2t,
                                                        const float* __restrict__ b1,
                                                        const float* __restrict__ b2,
                                                        float* __restrict__ emb) {
  __shared__ half_t sA[32 * KPAD];       // 6 KB   A strip (f16), 2 row groups
  __shared__ half_t sH[32 * 512];        // 32 KB  hidden chunk (f16)
  __shared__ float  sP[2 * 32 * 64];     // 16 KB  stage-2 partials
  const int tid = threadIdx.x, lane = tid & 31, w = tid >> 5;
  const int row0 = blockIdx.x * 32;
  const int colA = lane & 15, rsel = (lane >> 4) & 1;

  // Warm L2/WGP$ for this wave's first weight tiles.
  __builtin_prefetch(B1t + (size_t)((w * 4) * 16) * KPAD, 0, 3);
  __builtin_prefetch(B2t + (size_t)((w & 3) * 16) * EE, 0, 3);

  // Async global->LDS staging of the 32x96 f16 A strip (6144B = 384 x b128).
  for (int i = tid; i < 384; i += 256)
    async_copy_b128(Xh + (size_t)row0 * KPAD + i * 8, sA + i * 8);
  wait_async0();
  __syncthreads();

  const int nt2 = w & 3, kh = w >> 2;    // stage-2 wave roles
  v8f acc2_0 = {}, acc2_1 = {};

  for (int c = 0; c < 8; ++c) {
    // ---- stage 1: hidden cols [c*512, c*512+512) ----
#pragma unroll
    for (int t = 0; t < 4; ++t) {
      const int nloc = (w * 4 + t) * 16;       // 0..496
      const int n = c * 512 + nloc;
      v8f acc0 = {}, acc1 = {};
#pragma unroll
      for (int ks = 0; ks < 3; ++ks) {
        v16h b  = glb_bfrag(B1t, KPAD, n, ks * 32, lane);
        v16h a0 = lds_afrag(sA,            KPAD, ks * 32, lane);
        v16h a1 = lds_afrag(sA + 16 * KPAD, KPAD, ks * 32, lane);
        acc0 = wmma32(a0, b, acc0);
        acc1 = wmma32(a1, b, acc1);
      }
      const float bias = b1[n + colA];
#pragma unroll
      for (int r = 0; r < 8; ++r) {
        const int rr = r + (rsel ? 8 : 0);
        float v0 = acc0[r] + bias; v0 = v0 > 0.0f ? v0 : 0.0f;
        float v1 = acc1[r] + bias; v1 = v1 > 0.0f ? v1 : 0.0f;
        sH[rr * 512 + nloc + colA]        = (half_t)v0;
        sH[(16 + rr) * 512 + nloc + colA] = (half_t)v1;
      }
    }
    __syncthreads();
    // ---- stage 2: accumulate K chunk [c*512, c*512+512) ----
#pragma unroll
    for (int ks = 0; ks < 8; ++ks) {
      const int kloc = kh * 256 + ks * 32;
      v16h b  = glb_bfrag(B2t, EE, nt2 * 16, c * 512 + kloc, lane);
      v16h a0 = lds_afrag(sH,            512, kloc, lane);
      v16h a1 = lds_afrag(sH + 16 * 512, 512, kloc, lane);
      acc2_0 = wmma32(a0, b, acc2_0);
      acc2_1 = wmma32(a1, b, acc2_1);
    }
    __syncthreads();                            // sH reused by next chunk
  }
#pragma unroll
  for (int r = 0; r < 8; ++r) {
    const int rr = r + (rsel ? 8 : 0);
    sP[kh * 2048 + rr * 64        + nt2 * 16 + colA] = acc2_0[r];
    sP[kh * 2048 + (16 + rr) * 64 + nt2 * 16 + colA] = acc2_1[r];
  }
  __syncthreads();
  for (int i = tid; i < 2048; i += 256) {
    const int rr = i >> 6, col = i & 63;
    float v = sP[i] + sP[2048 + i] + b2[col];
    v = v > 0.0f ? v : 0.0f;
    emb[(size_t)(row0 + rr) * 64 + col] = v;
  }
}

// ---------------------------------------------------------------------------
// 5) Sequential 2-layer LSTM, persistent single workgroup, state in LDS.
//    Also emits progress[b][s] = sigmoid(h2 @ fc8.T + b8).
// ---------------------------------------------------------------------------
__global__ __launch_bounds__(256) void lstm_seq_kernel(
    const float* __restrict__ emb,
    const float* __restrict__ wih1, const float* __restrict__ whh1,
    const float* __restrict__ bih1, const float* __restrict__ bhh1,
    const float* __restrict__ wih2, const float* __restrict__ whh2,
    const float* __restrict__ bih2, const float* __restrict__ bhh2,
    const float* __restrict__ fc8w, const float* __restrict__ fc8b,
    float* __restrict__ H1, float* __restrict__ C1,
    float* __restrict__ H2, float* __restrict__ C2,
    float* __restrict__ prog) {
  __shared__ float x[16 * 64], h1[16 * 64], c1s[16 * 64];
  __shared__ float h2[16 * 32], c2s[16 * 32];
  __shared__ float g1[16 * 256], g2[16 * 128];
  const int tid = threadIdx.x;
  for (int i = tid; i < 1024; i += 256) { h1[i] = 0.0f; c1s[i] = 0.0f; }
  for (int i = tid; i < 512;  i += 256) { h2[i] = 0.0f; c2s[i] = 0.0f; }
  __syncthreads();

  for (int s = 0; s < SS; ++s) {
    for (int i = tid; i < 1024; i += 256) {
      const int b = i >> 6, k = i & 63;
      x[i] = emb[(size_t)(b * SS + s) * 64 + k];
    }
    __syncthreads();
    {   // layer-1 gates: thread tid owns gate row tid (0..255) for all 16 batches
      const float bias = bih1[tid] + bhh1[tid];
      const float* wi = wih1 + tid * 64;
      const float* wh = whh1 + tid * 64;
      for (int b = 0; b < 16; ++b) {
        float a = bias;
        for (int k = 0; k < 64; ++k) a += wi[k] * x[b * 64 + k] + wh[k] * h1[b * 64 + k];
        g1[b * 256 + tid] = a;
      }
    }
    __syncthreads();
    for (int i = tid; i < 1024; i += 256) {     // layer-1 cell update
      const int b = i >> 6, cc = i & 63;
      const float gi = g1[b * 256 + cc],       gf = g1[b * 256 + 64 + cc];
      const float gg = g1[b * 256 + 128 + cc], go = g1[b * 256 + 192 + cc];
      const float cn = sigf(gf) * c1s[i] + sigf(gi) * tanhf(gg);
      const float hn = sigf(go) * tanhf(cn);
      c1s[i] = cn; h1[i] = hn;
      H1[(size_t)s * 1024 + i] = hn; C1[(size_t)s * 1024 + i] = cn;
    }
    __syncthreads();
    for (int j = 0; j < 8; ++j) {               // layer-2 gates (2048 values)
      const int idx = tid + 256 * j, b = idx >> 7, gr = idx & 127;
      float a = bih2[gr] + bhh2[gr];
      const float* wi = wih2 + gr * 64;
      const float* wh = whh2 + gr * 32;
      for (int k = 0; k < 64; ++k) a += wi[k] * h1[b * 64 + k];
      for (int k = 0; k < 32; ++k) a += wh[k] * h2[b * 32 + k];
      g2[b * 128 + gr] = a;
    }
    __syncthreads();
    for (int i = tid; i < 512; i += 256) {      // layer-2 cell update
      const int b = i >> 5, cc = i & 31;
      const float gi = g2[b * 128 + cc],      gf = g2[b * 128 + 32 + cc];
      const float gg = g2[b * 128 + 64 + cc], go = g2[b * 128 + 96 + cc];
      const float cn = sigf(gf) * c2s[i] + sigf(gi) * tanhf(gg);
      const float hn = sigf(go) * tanhf(cn);
      c2s[i] = cn; h2[i] = hn;
      H2[(size_t)s * 512 + i] = hn; C2[(size_t)s * 512 + i] = cn;
    }
    __syncthreads();
    if (tid < 16) {                             // progress head
      float a = fc8b[0];
      for (int k = 0; k < 32; ++k) a += fc8w[k] * h2[tid * 32 + k];
      prog[tid * SS + s] = sigf(a);
    }
    __syncthreads();
  }
}

// ---------------------------------------------------------------------------
// 6) Forecast one-step cell, one block per (b,s) row; fully parallel.
// ---------------------------------------------------------------------------
__global__ __launch_bounds__(256) void forecast_cell_kernel(
    const float* __restrict__ emb,
    const float* __restrict__ H1, const float* __restrict__ C1,
    const float* __restrict__ H2, const float* __restrict__ C2,
    const float* __restrict__ wih1, const float* __restrict__ whh1,
    const float* __restrict__ bih1, const float* __restrict__ bhh1,
    const float* __restrict__ wih2, const float* __restrict__ whh2,
    const float* __restrict__ bih2, const float* __restrict__ bhh2,
    const float* __restrict__ fc8w, const float* __restrict__ fc8b,
    float* __restrict__ fprog) {
  __shared__ float xf[64], hp1[64], cp1[64], hp2[32], cp2[32];
  __shared__ float g1[256], fh1[64], g2[128], fh2[32];
  const int r = blockIdx.x, tid = threadIdx.x;
  const int b = r >> 9, s = r & 511;
  if (tid < 64) {
    xf[tid]  = emb[(size_t)(NFRM + r) * 64 + tid];      // femb rows
    hp1[tid] = H1[(size_t)s * 1024 + b * 64 + tid];
    cp1[tid] = C1[(size_t)s * 1024 + b * 64 + tid];
  } else if (tid < 96) {
    const int t = tid - 64;
    hp2[t] = H2[(size_t)s * 512 + b * 32 + t];
    cp2[t] = C2[(size_t)s * 512 + b * 32 + t];
  }
  __syncthreads();
  {   // layer-1 gates
    float a = bih1[tid] + bhh1[tid];
    const float* wi = wih1 + tid * 64;
    const float* wh = whh1 + tid * 64;
    for (int k = 0; k < 64; ++k) a += wi[k] * xf[k] + wh[k] * hp1[k];
    g1[tid] = a;
  }
  __syncthreads();
  if (tid < 64) {
    const float cn = sigf(g1[64 + tid]) * cp1[tid] + sigf(g1[tid]) * tanhf(g1[128 + tid]);
    fh1[tid] = sigf(g1[192 + tid]) * tanhf(cn);
  }
  __syncthreads();
  if (tid < 128) {
    float a = bih2[tid] + bhh2[tid];
    const float* wi = wih2 + tid * 64;
    const float* wh = whh2 + tid * 32;
    for (int k = 0; k < 64; ++k) a += wi[k] * fh1[k];
    for (int k = 0; k < 32; ++k) a += wh[k] * hp2[k];
    g2[tid] = a;
  }
  __syncthreads();
  if (tid < 32) {
    const float cn = sigf(g2[32 + tid]) * cp2[tid] + sigf(g2[tid]) * tanhf(g2[64 + tid]);
    fh2[tid] = sigf(g2[96 + tid]) * tanhf(cn);
  }
  __syncthreads();
  if (tid == 0) {
    float a = fc8b[0];
    for (int k = 0; k < 32; ++k) a += fc8w[k] * fh2[k];
    fprog[r] = sigf(a);
  }
}

// ---------------------------------------------------------------------------
// Host-side orchestration
// ---------------------------------------------------------------------------
extern "C" void kernel_launch(void* const* d_in, const int* in_sizes, int n_in,
                              void* d_out, int out_size, void* d_ws, size_t ws_size,
                              hipStream_t stream) {
  (void)in_sizes; (void)n_in; (void)out_size;
  const float* frames  = (const float*)d_in[0];
  const float* spp_w   = (const float*)d_in[1];
  const float* spp_b   = (const float*)d_in[2];
  const float* ffc_w   = (const float*)d_in[3];
  const float* ffc_b   = (const float*)d_in[4];
  const float* fc7_w   = (const float*)d_in[5];
  const float* fc7_b   = (const float*)d_in[6];
  const float* l1_wih  = (const float*)d_in[7];
  const float* l1_whh  = (const float*)d_in[8];
  const float* l1_bih  = (const float*)d_in[9];
  const float* l1_bhh  = (const float*)d_in[10];
  const float* l2_wih  = (const float*)d_in[11];
  const float* l2_whh  = (const float*)d_in[12];
  const float* l2_bih  = (const float*)d_in[13];
  const float* l2_bhh  = (const float*)d_in[14];
  const float* fc8_w   = (const float*)d_in[15];
  const float* fc8_b   = (const float*)d_in[16];

  float* out = (float*)d_out;
  float* prog    = out;                       // [16][512]
  float* fprog   = out + NFRM;                // [16][512]
  float* pooled  = out + 2 * NFRM;            // [8192][90]
  float* fpooled = out + 2 * NFRM + NFRM*90;  // [8192][90]

  // workspace layout (bytes); ~14.3 MB total
  char* ws = (char*)d_ws; (void)ws_size;
  half_t* Xh  = (half_t*)(ws);                                   // 16384*96 f16
  half_t* B1t = (half_t*)(ws + 3145728);                         // 4096*96 f16
  half_t* B2t = (half_t*)(ws + 3932160);                         // 64*4096 f16
  float*  emb = (float*)(ws + 4456448);                          // 16384*64 f32
  float*  H1  = (float*)(ws + 8650752);                          // 512*16*64
  float*  C1  = (float*)(ws + 10747904);
  float*  H2  = (float*)(ws + 12845056);                         // 512*16*32
  float*  C2  = (float*)(ws + 13893632);

  spp_pool_kernel<<<NFRM, 128, 0, stream>>>(frames, pooled);
  fc90_kernel<<<NFRM, 96, 0, stream>>>(pooled, ffc_w, ffc_b, fpooled);

  pack_x_kernel <<<(MROWS * KPAD + 255) / 256, 256, 0, stream>>>(pooled, fpooled, Xh);
  pack_w1_kernel<<<(EE * KPAD + 255) / 256,   256, 0, stream>>>(spp_w, B1t);
  pack_w2_kernel<<<(64 * EE + 255) / 256,     256, 0, stream>>>(fc7_w, B2t);

  fused_mlp_kernel<<<MROWS / 32, 256, 0, stream>>>(Xh, B1t, B2t, spp_b, fc7_b, emb);

  lstm_seq_kernel<<<1, 256, 0, stream>>>(emb,
      l1_wih, l1_whh, l1_bih, l1_bhh,
      l2_wih, l2_whh, l2_bih, l2_bhh,
      fc8_w, fc8_b, H1, C1, H2, C2, prog);

  forecast_cell_kernel<<<NFRM, 256, 0, stream>>>(emb, H1, C1, H2, C2,
      l1_wih, l1_whh, l1_bih, l1_bhh,
      l2_wih, l2_whh, l2_bih, l2_bhh,
      fc8_w, fc8_b, fprog);
}
